// Decoder_69209103008250
// MI455X (gfx1250) — matrix operand order repaired and checked by reading, not compiled, not run on hardware
//
#include <hip/hip_runtime.h>
#include <stdint.h>

typedef unsigned short u16;
typedef __attribute__((ext_vector_type(16))) __bf16 v16bf;
typedef __attribute__((ext_vector_type(8)))  float  v8f;

#define LROW 40   // LDS row stride in u16 (80 B): 16 lanes * b128 -> conflict-free banks

// ---------- helpers ----------
__device__ __forceinline__ u16 f2bf(float f) {
  union { float f; uint32_t u; } v; v.f = f;
  uint32_t u = v.u;
  uint32_t r = u + 0x7FFFu + ((u >> 16) & 1u);   // round-to-nearest-even
  return (u16)(r >> 16);
}

// A fragment: 16x32 bf16 (MxK). lane: m = lane&15, half = lane>>4.
// v[0..15] = A[m][half*16 .. +15]
__device__ __forceinline__ v16bf load_a_frag(const u16* p) {
  union { uint4 u[2]; v16bf v; } t;
  t.u[0] = *(const uint4*)p;
  t.u[1] = *(const uint4*)(p + 8);
  return t.v;
}
// B fragment: 32x16 bf16 (KxN), B[k][n] = W[n][k]. lane: n = lane&15, khalf = (lane>>4)*16.
// v[0..15] = 16 contiguous bf16 (32 B)
__device__ __forceinline__ v16bf load_b_frag(const u16* p) {
  union { uint4 u[2]; v16bf v; } t;
  t.u[0] = *(const uint4*)p;
  t.u[1] = *(const uint4*)(p + 8);
  return t.v;
}
__device__ __forceinline__ v8f wmma_bf16(v16bf a, v16bf b, v8f c) {
  return __builtin_amdgcn_wmma_f32_16x16x32_bf16(false, a, false, b, (short)0, c, false, false);
}

// ---------- f32 -> bf16 (optionally column-padded) ----------
__global__ void convert_pad_kernel(const float* __restrict__ src, u16* __restrict__ dst,
                                   int rows, int scols, int dcols) {
  size_t i = (size_t)blockIdx.x * blockDim.x + threadIdx.x;
  size_t total = (size_t)rows * dcols;
  if (i >= total) return;
  int r = (int)(i / dcols), c = (int)(i % dcols);
  dst[i] = (c < scols) ? f2bf(src[(size_t)r * scols + c]) : (u16)0;
}

__global__ void bias_comb_kernel(const float* __restrict__ a, const float* __restrict__ b,
                                 float* __restrict__ o) {
  int i = blockIdx.x * blockDim.x + threadIdx.x;
  if (i < 4096) o[i] = a[i] + b[i];
}

__global__ void zero_u32_kernel(uint32_t* p, int n) {
  int i = blockIdx.x * blockDim.x + threadIdx.x;
  if (i < n) p[i] = 0u;
}

// embedding gather into x[:, 1024..1343] (pad 1324..1343 with zero)
__global__ void embed_kernel(const int* __restrict__ targ, const float* __restrict__ emb,
                             u16* __restrict__ x) {
  int row = blockIdx.x;            // t*48 + b
  int tok = targ[row];
  int e = threadIdx.x;             // 0..319
  x[(size_t)row * 1344 + 1024 + e] = (e < 300) ? f2bf(emb[(size_t)tok * 300 + e]) : (u16)0;
}

// ---------- generic GEMM: C[M,N] = A[M,K](bf16) @ W[N,K]^T(bf16) + bias, act ----------
// 128 threads = 4 waves, each wave one 16-row M tile; blockIdx.x -> 64-col N strip.
// B chunk (64 rows x 32 k) staged via async Global->LDS, double-buffered (ASYNCcnt).
__global__ __launch_bounds__(128) void gemm_bf16_kernel(
    const u16* __restrict__ A, const u16* __restrict__ W,
    const float* __restrict__ bias,
    float* __restrict__ C, u16* __restrict__ Cbf,
    int M, int N, int K, int act) {
  __shared__ u16 shB[2][64 * LROW];

  const int tid  = threadIdx.x;
  const int lane = tid & 31;
  const int wave = tid >> 5;
  const int m0   = (blockIdx.y * 4 + wave) * 16;
  const int n0   = blockIdx.x * 64;

  const int hf  = (lane >> 4) & 1;
  const int mln = lane & 15;

  const int ar  = (m0 + mln < M) ? (m0 + mln) : 0;   // clamp (wave may be fully OOB)
  const u16* Ab = A + (size_t)ar * K + hf * 16;

  // staging: each of the 128 threads copies the first 16 B (b128) of one 32-byte half-row per chunk
  const int srow = tid >> 1;              // 0..63
  const int scol = (tid & 1) * 16;        // 0 or 16 elements
  const u16* Wsrc = W + (size_t)(n0 + srow) * K + scol;
  uint32_t lds0 = (uint32_t)(uintptr_t)&shB[0][srow * LROW + scol];
  uint32_t lds1 = (uint32_t)(uintptr_t)&shB[1][srow * LROW + scol];

#define STAGE(bufoff, kk)                                              \
  asm volatile("global_load_async_to_lds_b128 %0, %1, off"             \
               :: "v"(bufoff), "v"(Wsrc + (kk)) : "memory")

  v8f acc0 = {}, acc1 = {}, acc2 = {}, acc3 = {};

  int cur = 0;
  STAGE(lds0, 0);
  for (int k = 0; k < K; k += 32) {
    if (k + 32 < K) {
      STAGE(cur ? lds0 : lds1, k + 32);                 // prefetch next chunk
      asm volatile("s_wait_asynccnt 1" ::: "memory");   // current chunk landed
    } else {
      asm volatile("s_wait_asynccnt 0" ::: "memory");
    }
    __syncthreads();

    v16bf af = load_a_frag(Ab + k);
    const u16* bb = &shB[cur][0];
    v16bf b0 = load_b_frag(bb + ( 0 + mln) * LROW + hf * 16);
    v16bf b1 = load_b_frag(bb + (16 + mln) * LROW + hf * 16);
    v16bf b2 = load_b_frag(bb + (32 + mln) * LROW + hf * 16);
    v16bf b3 = load_b_frag(bb + (48 + mln) * LROW + hf * 16);
    acc0 = wmma_bf16(af, b0, acc0);
    acc1 = wmma_bf16(af, b1, acc1);
    acc2 = wmma_bf16(af, b2, acc2);
    acc3 = wmma_bf16(af, b3, acc3);
    __builtin_prefetch((const void*)(Ab + k + 256), 0, 1);

    __syncthreads();          // all waves done reading before next overwrite
    cur ^= 1;
  }
#undef STAGE

#pragma unroll
  for (int sub = 0; sub < 4; ++sub) {
    v8f acc = (sub == 0) ? acc0 : (sub == 1) ? acc1 : (sub == 2) ? acc2 : acc3;
    int n = n0 + sub * 16 + mln;
    float bv = bias ? bias[n] : 0.0f;
#pragma unroll
    for (int j = 0; j < 8; ++j) {
      int m = m0 + hf * 8 + j;
      if (m < M) {
        float v = acc[j] + bv;
        if (act == 1) v = tanhf(v);
        if (C)   __builtin_nontemporal_store(v, &C[(size_t)m * N + n]);
        if (Cbf) Cbf[(size_t)m * N + n] = f2bf(v);
      }
    }
  }
}

// ---------- attention pieces ----------
// sc[s,b] = dot(enc[s,b,:], ht[b,:]); one wave per (s,b)
__global__ void score_kernel(const float* __restrict__ enc, const float* __restrict__ ht,
                             float* __restrict__ sc) {
  int idx  = blockIdx.x * (blockDim.x >> 5) + (threadIdx.x >> 5);
  int lane = threadIdx.x & 31;
  if (idx >= 64 * 48) return;
  int s = idx / 48, b = idx % 48;
  const float* e = enc + ((size_t)s * 48 + b) * 1024;
  const float* h = ht + (size_t)b * 1024;
  float acc = 0.0f;
  for (int k = lane; k < 1024; k += 32) acc += e[k] * h[k];
  for (int off = 16; off > 0; off >>= 1) acc += __shfl_down(acc, off, 32);
  if (lane == 0) sc[idx] = acc;
}

// log_softmax over batch dim (axis=1), one block per s
__global__ void logsoftmax_kernel(const float* __restrict__ sc, float* __restrict__ a) {
  __shared__ float sh[64];
  int s = blockIdx.x, tid = threadIdx.x;
  float v = (tid < 48) ? sc[s * 48 + tid] : -INFINITY;
  sh[tid] = v; __syncthreads();
  for (int off = 32; off > 0; off >>= 1) { if (tid < off) sh[tid] = fmaxf(sh[tid], sh[tid + off]); __syncthreads(); }
  float mx = sh[0]; __syncthreads();
  float e = (tid < 48) ? __expf(v - mx) : 0.0f;
  sh[tid] = e; __syncthreads();
  for (int off = 32; off > 0; off >>= 1) { if (tid < off) sh[tid] += sh[tid + off]; __syncthreads(); }
  float lse = mx + __logf(sh[0]);
  if (tid < 48) a[s * 48 + tid] = v - lse;
}

// s_vec[b,d] = sum_s a[s,b]*enc[s,b,d]; write concat = [bf16(s_vec) | hidden_bf16]
__global__ void attn_ctx_kernel(const float* __restrict__ a, const float* __restrict__ enc,
                                const u16* __restrict__ hid_bf, u16* __restrict__ concat) {
  int b = blockIdx.x;
  int d = blockIdx.y * 256 + threadIdx.x;
  float acc = 0.0f;
  for (int s = 0; s < 64; ++s) acc += a[s * 48 + b] * enc[((size_t)s * 48 + b) * 1024 + d];
  concat[(size_t)b * 2048 + d]        = f2bf(acc);
  concat[(size_t)b * 2048 + 1024 + d] = hid_bf[(size_t)b * 1024 + d];
}

// broadcast ctx_bf16 into x[:, 0..1023] for every timestep
__global__ void bcast_ctx_kernel(const u16* __restrict__ ctx, u16* __restrict__ x) {
  int row = blockIdx.x;      // 0..3071
  int b = row % 48;
  for (int d = threadIdx.x; d < 1024; d += blockDim.x)
    x[(size_t)row * 1344 + d] = ctx[(size_t)b * 1024 + d];
}

// ---------- LSTM recurrent step ----------
// g = xg[t] + h @ w_hh^T, gates, cell update. Block: 3 waves (M=48), grid.x: D/16 tiles;
// each block owns cols d0..d0+15 of ALL FOUR gates -> in-register gate math on accumulators.
__global__ void lstm_step_kernel(const u16* __restrict__ h_in, const u16* __restrict__ Whh,
                                 const float* __restrict__ xg, float* __restrict__ c,
                                 u16* __restrict__ h_out, u16* __restrict__ hs, int t) {
  const int lane = threadIdx.x & 31;
  const int wave = threadIdx.x >> 5;     // 0..2 -> m tile
  const int m0   = wave * 16;
  const int d0   = blockIdx.x * 16;      // column tile in D
  const int hf   = (lane >> 4) & 1;
  const int mln  = lane & 15;

  const u16* Ab = h_in + (size_t)(m0 + mln) * 1024 + hf * 16;
  const u16* W0 = Whh + (size_t)(0 * 1024 + d0 + mln) * 1024 + hf * 16;
  const u16* W1 = Whh + (size_t)(1 * 1024 + d0 + mln) * 1024 + hf * 16;
  const u16* W2 = Whh + (size_t)(2 * 1024 + d0 + mln) * 1024 + hf * 16;
  const u16* W3 = Whh + (size_t)(3 * 1024 + d0 + mln) * 1024 + hf * 16;

  v8f ai = {}, af_ = {}, ag = {}, ao = {};
  for (int k = 0; k < 1024; k += 32) {
    v16bf a = load_a_frag(Ab + k);
    ai  = wmma_bf16(a, load_b_frag(W0 + k), ai);
    af_ = wmma_bf16(a, load_b_frag(W1 + k), af_);
    ag  = wmma_bf16(a, load_b_frag(W2 + k), ag);
    ao  = wmma_bf16(a, load_b_frag(W3 + k), ao);
  }

  const int n = d0 + mln;
#pragma unroll
  for (int j = 0; j < 8; ++j) {
    int b = m0 + hf * 8 + j;                         // batch row 0..47
    size_t xr = ((size_t)t * 48 + b) * 4096;
    float gi = ai[j]  + xg[xr + 0 * 1024 + n];
    float gf = af_[j] + xg[xr + 1 * 1024 + n];
    float gg = ag[j]  + xg[xr + 2 * 1024 + n];
    float go = ao[j]  + xg[xr + 3 * 1024 + n];
    float si = 1.0f / (1.0f + __expf(-gi));
    float sf = 1.0f / (1.0f + __expf(-gf));
    float so = 1.0f / (1.0f + __expf(-go));
    float cv = sf * c[(size_t)b * 1024 + n] + si * tanhf(gg);
    c[(size_t)b * 1024 + n] = cv;
    u16 hb = f2bf(so * tanhf(cv));
    h_out[(size_t)b * 1024 + n] = hb;
    hs[((size_t)t * 48 + b) * 1024 + n] = hb;
  }
}

// ---------- host launcher ----------
extern "C" void kernel_launch(void* const* d_in, const int* in_sizes, int n_in,
                              void* d_out, int out_size, void* d_ws, size_t ws_size,
                              hipStream_t stream) {
  (void)in_sizes; (void)n_in; (void)out_size; (void)ws_size;
  const int*   targ     = (const int*)d_in[0];
  const float* enc      = (const float*)d_in[1];
  const float* emb      = (const float*)d_in[2];
  const float* attin_w  = (const float*)d_in[3];
  const float* attin_b  = (const float*)d_in[4];
  const float* attout_w = (const float*)d_in[5];
  const float* attout_b = (const float*)d_in[6];
  const float* gen_w    = (const float*)d_in[7];
  const float* gen_b    = (const float*)d_in[8];
  const float* hidden   = (const float*)d_in[9];
  const float* w_ih     = (const float*)d_in[10];
  const float* w_hh     = (const float*)d_in[11];
  const float* b_ih     = (const float*)d_in[12];
  const float* b_hh     = (const float*)d_in[13];
  float* out = (float*)d_out;

  char* base = (char*)d_ws;
  size_t off = 0;
  auto alloc = [&](size_t bytes) -> void* {
    void* p = base + off; off += (bytes + 255) & ~(size_t)255; return p;
  };
  u16*   hid_bf    = (u16*)  alloc((size_t)48 * 1024 * 2);
  u16*   attinw_bf = (u16*)  alloc((size_t)1024 * 1024 * 2);
  u16*   attoutw_bf= (u16*)  alloc((size_t)1024 * 2048 * 2);
  u16*   whh_bf    = (u16*)  alloc((size_t)4096 * 1024 * 2);
  u16*   genw_bf   = (u16*)  alloc((size_t)32000 * 1024 * 2);
  u16*   wih_bf    = (u16*)  alloc((size_t)4096 * 1344 * 2);
  u16*   xbuf      = (u16*)  alloc((size_t)3072 * 1344 * 2);
  float* ht        = (float*)alloc((size_t)48 * 1024 * 4);
  float* sc        = (float*)alloc((size_t)64 * 48 * 4);
  float* attp      = (float*)alloc((size_t)64 * 48 * 4);
  u16*   concat    = (u16*)  alloc((size_t)48 * 2048 * 2);
  u16*   ctx_bf    = (u16*)  alloc((size_t)48 * 1024 * 2);
  float* biasc     = (float*)alloc((size_t)4096 * 4);
  float* xg        = (float*)alloc((size_t)3072 * 4096 * 4);
  float* cbuf      = (float*)alloc((size_t)48 * 1024 * 4);
  u16*   hping     = (u16*)  alloc((size_t)48 * 1024 * 2);
  u16*   hpong     = (u16*)  alloc((size_t)48 * 1024 * 2);
  u16*   hs        = (u16*)  alloc((size_t)3072 * 1024 * 2);

  auto cvt = [&](const float* s, u16* d, int rows, int scols, int dcols) {
    size_t total = (size_t)rows * dcols;
    convert_pad_kernel<<<(unsigned)((total + 255) / 256), 256, 0, stream>>>(s, d, rows, scols, dcols);
  };
  cvt(hidden,   hid_bf,    48,   1024, 1024);
  cvt(attin_w,  attinw_bf, 1024, 1024, 1024);
  cvt(attout_w, attoutw_bf,1024, 2048, 2048);
  cvt(w_hh,     whh_bf,    4096, 1024, 1024);
  cvt(gen_w,    genw_bf,   32000,1024, 1024);
  cvt(w_ih,     wih_bf,    4096, 1324, 1344);

  bias_comb_kernel<<<16, 256, 0, stream>>>(b_ih, b_hh, biasc);
  embed_kernel<<<3072, 320, 0, stream>>>(targ, emb, xbuf);

  // ht = hidden @ attin_w^T + attin_b    (M=48, N=1024, K=1024)
  gemm_bf16_kernel<<<dim3(16, 1), 128, 0, stream>>>(hid_bf, attinw_bf, attin_b,
                                                    ht, nullptr, 48, 1024, 1024, 0);
  score_kernel<<<768, 128, 0, stream>>>(enc, ht, sc);
  logsoftmax_kernel<<<64, 64, 0, stream>>>(sc, attp);
  attn_ctx_kernel<<<dim3(48, 4), 256, 0, stream>>>(attp, enc, hid_bf, concat);
  // ctx = tanh(concat @ attout_w^T + attout_b)   (M=48, N=1024, K=2048)
  gemm_bf16_kernel<<<dim3(16, 1), 128, 0, stream>>>(concat, attoutw_bf, attout_b,
                                                    nullptr, ctx_bf, 48, 1024, 2048, 1);
  bcast_ctx_kernel<<<3072, 256, 0, stream>>>(ctx_bf, xbuf);
  // xg = x @ w_ih^T + (b_ih + b_hh)   (M=3072, N=4096, K=1344)
  gemm_bf16_kernel<<<dim3(64, 48), 128, 0, stream>>>(xbuf, wih_bf, biasc,
                                                     xg, nullptr, 3072, 4096, 1344, 0);
  // zero init h, c
  zero_u32_kernel<<<(49152 + 255) / 256, 256, 0, stream>>>((uint32_t*)cbuf, 49152);
  zero_u32_kernel<<<(24576 + 255) / 256, 256, 0, stream>>>((uint32_t*)hping, 24576);

  for (int t = 0; t < 64; ++t) {
    const u16* hin = (t & 1) ? hpong : hping;
    u16* hout      = (t & 1) ? hping : hpong;
    lstm_step_kernel<<<64, 96, 0, stream>>>(hin, whh_bf, xg, cbuf, hout, hs, t);
  }
  // out = hs @ gen_w^T + gen_b   (M=3072, N=32000, K=1024)
  gemm_bf16_kernel<<<dim3(500, 48), 128, 0, stream>>>(hs, genw_bf, gen_b,
                                                      out, nullptr, 3072, 32000, 1024, 0);
}
